// ExtraExpert_49555332661870
// MI455X (gfx1250) — compile-verified, matrix operand order checked by reading, not run on hardware
//
#include <hip/hip_runtime.h>

#define DIM   1024
#define HID   2816
#define E_TOT 9

#define TM   128
#define TN   64
#define KS   32
#define LSTR 40   // LDS row stride (bf16 elems), 80B: keeps 16B alignment, dodges bank conflicts

typedef __bf16 v16bf __attribute__((ext_vector_type(16)));
typedef __bf16 v8bf  __attribute__((ext_vector_type(8)));
typedef float  v8f   __attribute__((ext_vector_type(8)));

__device__ __forceinline__ __bf16 f2bf(float f) {
  union { float f; unsigned u; } v; v.f = f;
  unsigned r = v.u + 0x7FFFu + ((v.u >> 16) & 1u);   // round-to-nearest-even
  union { unsigned short s; __bf16 b; } o; o.s = (unsigned short)(r >> 16);
  return o.b;
}

// ---- CDNA5 async DMA: global -> LDS, 16B per lane, tracked on ASYNCcnt ----
__device__ __forceinline__ void async_b128(unsigned lds_byte_off, const void* gptr) {
  unsigned long long ga = (unsigned long long)gptr;
  asm volatile("global_load_async_to_lds_b128 %0, %1, off"
               :: "v"(lds_byte_off), "v"(ga) : "memory");
}
__device__ __forceinline__ void wait_async0() {
#if __has_builtin(__builtin_amdgcn_s_wait_asynccnt)
  __builtin_amdgcn_s_wait_asynccnt(0);
#else
  asm volatile("s_wait_asynccnt 0" ::: "memory");
#endif
}
// generic LDS pointer carries the wave-relative LDS byte address in its low 32 bits
__device__ __forceinline__ unsigned lds_off(const void* p) {
  return (unsigned)(size_t)p;
}

// Load one 16x32 bf16 WMMA operand fragment for this lane from an LDS tile.
// ISA layout: lane l -> row (l&15); halves 0..7 = K[koff..koff+7], halves 8..15 = K[koff+16..koff+23],
// koff = (l>>4)*8.  Two 16B reads.
__device__ __forceinline__ v16bf ld_frag(const __bf16* p) {
  v8bf lo = *(const v8bf*)p;
  v8bf hi = *(const v8bf*)(p + 16);
  return __builtin_shufflevector(lo, hi, 0,1,2,3,4,5,6,7,8,9,10,11,12,13,14,15);
}

// ---------------- fp32 -> bf16 conversion ----------------
__global__ __launch_bounds__(256) void cvt_k(const float* __restrict__ in,
                                             __bf16* __restrict__ out, size_t n) {
  size_t i = (size_t)blockIdx.x * blockDim.x + threadIdx.x;
  if (i < n) out[i] = f2bf(in[i]);
}

// ---------------- router: sigmoid scores, top-2, normalized combine weights ----------------
__global__ __launch_bounds__(256) void router_k(const float* __restrict__ x,
                                                const float* __restrict__ gw,   // [8, DIM]
                                                const float* __restrict__ ngw,  // [1, DIM]
                                                const float* __restrict__ gbias,// [1]
                                                float* __restrict__ cw,         // [T, 9]
                                                int T) {
  const int t = blockIdx.x * 8 + (threadIdx.x >> 5);
  const int lane = threadIdx.x & 31;
  if (t >= T) return;
  float acc[E_TOT];
#pragma unroll
  for (int e = 0; e < E_TOT; ++e) acc[e] = 0.f;
  for (int d = lane; d < DIM; d += 32) {
    const float xv = x[(size_t)t * DIM + d];
#pragma unroll
    for (int e = 0; e < 8; ++e) acc[e] += xv * gw[e * DIM + d];
    acc[8] += xv * ngw[d];
  }
#pragma unroll
  for (int e = 0; e < E_TOT; ++e)
    for (int off = 16; off > 0; off >>= 1)
      acc[e] += __shfl_xor(acc[e], off, 32);
  if (lane == 0) {
    float s[E_TOT];
#pragma unroll
    for (int e = 0; e < E_TOT; ++e) {
      float z = acc[e] + (e == 8 ? gbias[0] : 0.f);
      s[e] = __builtin_amdgcn_rcpf(1.f + __expf(-z));   // sigmoid via fast rcp
    }
    int i0 = 0;
#pragma unroll
    for (int e = 1; e < E_TOT; ++e) if (s[e] > s[i0]) i0 = e;
    int i1 = (i0 == 0) ? 1 : 0;
#pragma unroll
    for (int e = 0; e < E_TOT; ++e) if (e != i0 && s[e] > s[i1]) i1 = e;
    const float rdenom = __builtin_amdgcn_rcpf(s[i0] + s[i1] + 1e-20f);
#pragma unroll
    for (int e = 0; e < E_TOT; ++e) cw[(size_t)t * E_TOT + e] = 0.f;
    cw[(size_t)t * E_TOT + i0] = s[i0] * rdenom;
    cw[(size_t)t * E_TOT + i1] = s[i1] * rdenom;
  }
}

// ---------------- fused GEMM1 + GEMM3 + SiLU-mul:  Hb = silu(X W1^T) * (X W3^T) ----------------
// One expert per launch.  Block tile: 128 tokens x 64 hid.  Double-buffered async-DMA staging.
__global__ __launch_bounds__(256) void gemm13_k(const __bf16* __restrict__ xb,  // [T, DIM]
                                                const __bf16* __restrict__ w1,  // [HID, DIM]
                                                const __bf16* __restrict__ w3,  // [HID, DIM]
                                                __bf16* __restrict__ hb) {      // [T, HID]
  __shared__ __align__(16) __bf16 lA [2][TM * LSTR];
  __shared__ __align__(16) __bf16 lB1[2][TN * LSTR];
  __shared__ __align__(16) __bf16 lB3[2][TN * LSTR];

  const int m0 = blockIdx.x * TM;
  const int n0 = blockIdx.y * TN;
  const int tid  = threadIdx.x;
  const int lane = tid & 31;
  const int wave = tid >> 5;
  const int wm = (wave >> 1) * 32;   // 4 waves along M
  const int wn = (wave & 1) * 32;    // 2 waves along N

  // cooperative staging: A = 128x32 (32B/thread), B = 2 matrices x 64x32 (32B/thread)
  const int arow = tid >> 1,  acol = (tid & 1) * 16;
  const int bsel = tid >> 7,  brow = (tid & 127) >> 1, bcol = (tid & 1) * 16;
  const __bf16* gA = &xb[(size_t)(m0 + arow) * DIM + acol];
  const __bf16* gB = &(bsel ? w3 : w1)[(size_t)(n0 + brow) * DIM + bcol];

  unsigned aoff[2], boff[2];
#pragma unroll
  for (int b = 0; b < 2; ++b) {
    aoff[b] = lds_off(&lA[b][arow * LSTR + acol]);
    boff[b] = lds_off(bsel ? &lB3[b][brow * LSTR + bcol] : &lB1[b][brow * LSTR + bcol]);
  }

  const int mlane = lane & 15;
  const int koff  = (lane >> 4) * 8;

  v8f acc1[2][2] = {};
  v8f acc3[2][2] = {};

  // prime buffer 0
  async_b128(aoff[0],      gA);
  async_b128(aoff[0] + 16, gA + 8);
  async_b128(boff[0],      gB);
  async_b128(boff[0] + 16, gB + 8);

  int buf = 0;
  for (int k0 = 0; k0 < DIM; k0 += KS, buf ^= 1) {
    wait_async0();        // tile[buf] landed (this wave's DMAs)
    __syncthreads();      // all waves' DMAs landed; prior reads of tile[buf^1] retired
    if (k0 + KS < DIM) {  // kick DMA for next tile into the other buffer
      const int nb = buf ^ 1, k1 = k0 + KS;
      async_b128(aoff[nb],      gA + k1);
      async_b128(aoff[nb] + 16, gA + k1 + 8);
      async_b128(boff[nb],      gB + k1);
      async_b128(boff[nb] + 16, gB + k1 + 8);
    }

    v16bf af[2], f1[2], f3[2];
#pragma unroll
    for (int mi = 0; mi < 2; ++mi)
      af[mi] = ld_frag(&lA[buf][(wm + mi * 16 + mlane) * LSTR + koff]);
#pragma unroll
    for (int ni = 0; ni < 2; ++ni) {
      f1[ni] = ld_frag(&lB1[buf][(wn + ni * 16 + mlane) * LSTR + koff]);
      f3[ni] = ld_frag(&lB3[buf][(wn + ni * 16 + mlane) * LSTR + koff]);
    }
#pragma unroll
    for (int mi = 0; mi < 2; ++mi)
#pragma unroll
      for (int ni = 0; ni < 2; ++ni) {
        acc1[mi][ni] = __builtin_amdgcn_wmma_f32_16x16x32_bf16(
            false, af[mi], false, f1[ni], (short)0, acc1[mi][ni], false, false);
        acc3[mi][ni] = __builtin_amdgcn_wmma_f32_16x16x32_bf16(
            false, af[mi], false, f3[ni], (short)0, acc3[mi][ni], false, false);
      }
  }

  // epilogue: h = silu(g) * u  (fast rcp, no IEEE divide), store bf16
  const int drow = (lane >> 4) * 8;
  const int dcol = lane & 15;
#pragma unroll
  for (int mi = 0; mi < 2; ++mi)
#pragma unroll
    for (int ni = 0; ni < 2; ++ni)
#pragma unroll
      for (int i = 0; i < 8; ++i) {
        float g = acc1[mi][ni][i];
        float u = acc3[mi][ni][i];
        float h = g * __builtin_amdgcn_rcpf(1.f + __expf(-g)) * u;
        hb[(size_t)(m0 + wm + mi * 16 + drow + i) * HID + (n0 + wn + ni * 16 + dcol)] = f2bf(h);
      }
}

// ---------------- GEMM2:  out (+)= cw[:,e] * (Hb W2^T) ----------------
__global__ __launch_bounds__(256) void gemm2_k(const __bf16* __restrict__ hb,  // [T, HID]
                                               const __bf16* __restrict__ w2,  // [DIM, HID]
                                               const float* __restrict__ cw,   // [T, 9]
                                               int e, float* __restrict__ out, // [T, DIM]
                                               int first) {
  __shared__ __align__(16) __bf16 lA[2][TM * LSTR];
  __shared__ __align__(16) __bf16 lB[2][TN * LSTR];

  const int m0 = blockIdx.x * TM;
  const int n0 = blockIdx.y * TN;
  const int tid  = threadIdx.x;
  const int lane = tid & 31;
  const int wave = tid >> 5;
  const int wm = (wave >> 1) * 32;
  const int wn = (wave & 1) * 32;

  const int arow = tid >> 1, acol = (tid & 1) * 16;
  const int brow = tid >> 2, bcol = (tid & 3) * 8;
  const __bf16* gA = &hb[(size_t)(m0 + arow) * HID + acol];
  const __bf16* gB = &w2[(size_t)(n0 + brow) * HID + bcol];

  unsigned aoff[2], boff[2];
#pragma unroll
  for (int b = 0; b < 2; ++b) {
    aoff[b] = lds_off(&lA[b][arow * LSTR + acol]);
    boff[b] = lds_off(&lB[b][brow * LSTR + bcol]);
  }

  const int mlane = lane & 15;
  const int koff  = (lane >> 4) * 8;

  v8f acc[2][2] = {};

  async_b128(aoff[0],      gA);
  async_b128(aoff[0] + 16, gA + 8);
  async_b128(boff[0],      gB);        // B: 8 bf16/thread -> one b128 covers it

  int buf = 0;
  for (int k0 = 0; k0 < HID; k0 += KS, buf ^= 1) {
    wait_async0();
    __syncthreads();
    if (k0 + KS < HID) {
      const int nb = buf ^ 1, k1 = k0 + KS;
      async_b128(aoff[nb],      gA + k1);
      async_b128(aoff[nb] + 16, gA + k1 + 8);
      async_b128(boff[nb],      gB + k1);
    }

    v16bf af[2], bfr[2];
#pragma unroll
    for (int mi = 0; mi < 2; ++mi)
      af[mi] = ld_frag(&lA[buf][(wm + mi * 16 + mlane) * LSTR + koff]);
#pragma unroll
    for (int ni = 0; ni < 2; ++ni)
      bfr[ni] = ld_frag(&lB[buf][(wn + ni * 16 + mlane) * LSTR + koff]);
#pragma unroll
    for (int mi = 0; mi < 2; ++mi)
#pragma unroll
      for (int ni = 0; ni < 2; ++ni)
        acc[mi][ni] = __builtin_amdgcn_wmma_f32_16x16x32_bf16(
            false, af[mi], false, bfr[ni], (short)0, acc[mi][ni], false, false);
  }

  const int drow = (lane >> 4) * 8;
  const int dcol = lane & 15;
#pragma unroll
  for (int mi = 0; mi < 2; ++mi)
#pragma unroll
    for (int ni = 0; ni < 2; ++ni)
#pragma unroll
      for (int i = 0; i < 8; ++i) {
        const int row = m0 + wm + mi * 16 + drow + i;
        const int col = n0 + wn + ni * 16 + dcol;
        const float v = cw[(size_t)row * E_TOT + e] * acc[mi][ni][i];
        const size_t idx = (size_t)row * DIM + col;
        if (first) out[idx] = v;
        else       out[idx] += v;
      }
}

extern "C" void kernel_launch(void* const* d_in, const int* in_sizes, int n_in,
                              void* d_out, int out_size, void* d_ws, size_t ws_size,
                              hipStream_t stream) {
  const float* x   = (const float*)d_in[0];
  const float* w1  = (const float*)d_in[1];
  const float* w2  = (const float*)d_in[2];
  const float* w3  = (const float*)d_in[3];
  const float* nw1 = (const float*)d_in[4];
  const float* nw2 = (const float*)d_in[5];
  const float* nw3 = (const float*)d_in[6];
  const float* gw  = (const float*)d_in[7];
  const float* ngw = (const float*)d_in[8];
  const float* gb  = (const float*)d_in[9];
  float* out = (float*)d_out;
  const int T = in_sizes[0] / DIM;   // 4096

  // workspace layout (all 16B aligned)
  char* ws = (char*)d_ws;
  size_t off = 0;
  const size_t WSZ = (size_t)HID * DIM;                 // elems per expert matrix
  __bf16* xb   = (__bf16*)(ws + off); off += (size_t)T * DIM * 2;
  __bf16* wb1  = (__bf16*)(ws + off); off += (size_t)E_TOT * WSZ * 2;
  __bf16* wb3  = (__bf16*)(ws + off); off += (size_t)E_TOT * WSZ * 2;
  __bf16* wb2  = (__bf16*)(ws + off); off += (size_t)E_TOT * WSZ * 2;
  __bf16* hbuf = (__bf16*)(ws + off); off += (size_t)T * HID * 2;
  float*  cw   = (float*)(ws + off);  off += (size_t)T * E_TOT * 4;

  auto cvt = [&](const float* src, __bf16* dst, size_t n) {
    cvt_k<<<(unsigned)((n + 255) / 256), 256, 0, stream>>>(src, dst, n);
  };
  cvt(x,   xb,            (size_t)T * DIM);
  cvt(w1,  wb1,           8 * WSZ);
  cvt(nw1, wb1 + 8 * WSZ, WSZ);
  cvt(w3,  wb3,           8 * WSZ);
  cvt(nw3, wb3 + 8 * WSZ, WSZ);
  cvt(w2,  wb2,           8 * WSZ);
  cvt(nw2, wb2 + 8 * WSZ, WSZ);

  router_k<<<T / 8, 256, 0, stream>>>(x, gw, ngw, gb, cw, T);

  dim3 g13(T / TM, HID / TN);   // 32 x 44
  dim3 g2 (T / TM, DIM / TN);   // 32 x 16
  for (int e = 0; e < E_TOT; ++e) {
    gemm13_k<<<g13, 256, 0, stream>>>(xb, wb1 + (size_t)e * WSZ, wb3 + (size_t)e * WSZ, hbuf);
    gemm2_k <<<g2,  256, 0, stream>>>(hbuf, wb2 + (size_t)e * WSZ, cw, e, out, e == 0 ? 1 : 0);
  }
}
